// get_model_16750372455079
// MI455X (gfx1250) — compile-verified, matrix operand order checked by reading, not run on hardware
//
#include <hip/hip_runtime.h>
#include <hip/hip_bf16.h>
#include <cstdint>
#include <cstddef>

#define EPSV 1e-5f
#define BATCH 8
#define NPTS 2048

typedef __bf16 bf16;
typedef __attribute__((ext_vector_type(16))) __bf16 v16bf;
typedef __attribute__((ext_vector_type(8)))  __bf16 v8bf;
typedef __attribute__((ext_vector_type(8)))  float  v8f;

// ---------------------------------------------------------------------------
// WMMA helpers (CDNA5: D = A(16x32 bf16) * B(32x16 bf16) + C(16x16 f32))
// ---------------------------------------------------------------------------
static __device__ inline v8f wmma_bf16(v16bf a, v16bf b, v8f c) {
  return __builtin_amdgcn_wmma_f32_16x16x32_bf16(false, a, false, b, (short)0, c,
                                                 false, false);
}

static __device__ inline v16bf join_frag(v8bf lo, v8bf hi) {
  v16bf a;
#pragma unroll
  for (int i = 0; i < 8; ++i) { a[i] = lo[i]; a[i + 8] = hi[i]; }
  return a;
}

// Software-pipelined K-loop GEMM core: A from a row-major bf16 buffer
// (LDS or global; two b128 loads per fragment), B from pre-packed weights
// (one 32-byte aligned vector load per lane). Fragments for kt+1 are issued
// before the WMMA for kt so the per-WMMA wait only covers in-flight loads.
static __device__ inline v8f gemm_core(const bf16* __restrict__ pa0,
                                       const bf16* __restrict__ pb0,
                                       size_t bstep, int ktiles) {
  const bf16* pa = pa0;
  const bf16* pb = pb0;
  v8bf  alo = *(const v8bf*)pa;
  v8bf  ahi = *(const v8bf*)(pa + 16);
  v16bf bfr = *(const v16bf*)pb;
  v8f acc = {};
  for (int kt = 0; kt < ktiles - 1; ++kt) {
    v8bf  nlo = *(const v8bf*)(pa + 32);
    v8bf  nhi = *(const v8bf*)(pa + 48);
    v16bf nb  = *(const v16bf*)(pb + bstep);
    acc = wmma_bf16(join_frag(alo, ahi), bfr, acc);
    alo = nlo; ahi = nhi; bfr = nb;
    pa += 32; pb += bstep;
  }
  acc = wmma_bf16(join_frag(alo, ahi), bfr, acc);
  return acc;
}

// ---------------------------------------------------------------------------
// weight pre-pack: f32 (Kd x Nn) row-major -> bf16 fragment layout, zero-pad
// ---------------------------------------------------------------------------
__global__ void pack_w_kernel(const float* __restrict__ W, int Kd, int Nn,
                              bf16* __restrict__ out, int ktiles, int ntiles) {
  int i = blockIdx.x * blockDim.x + threadIdx.x;
  int total = ktiles * ntiles * 512;
  if (i >= total) return;
  int e    = i & 15;
  int lane = (i >> 4) & 31;
  int t    = i >> 9;
  int kt = t / ntiles, nt = t % ntiles;
  int ii = e >> 1;
  int k = kt * 32 + ((ii < 4) ? 0 : 16) + ((lane < 16) ? 0 : 8) + 2 * (ii & 3) +
          (e & 1);
  int n = nt * 16 + (lane & 15);
  float v = (k < Kd && n < Nn) ? W[(size_t)k * Nn + n] : 0.f;
  out[i] = (bf16)v;
}

// ---------------------------------------------------------------------------
// 1) coordinate alignment:  coords = P[:,:, :3] @ (U * sign), feats copied
// ---------------------------------------------------------------------------
__global__ void align_kernel(const float* __restrict__ P, const float* __restrict__ U,
                             float sx, float sy, float sz, float* __restrict__ out) {
  int i = blockIdx.x * blockDim.x + threadIdx.x;
  if (i >= BATCH * NPTS) return;
  int b = i / NPTS;
  const float* p = P + (size_t)i * 6;
  const float* u = U + b * 9;
  float x = p[0], y = p[1], z = p[2];
  float* o = out + (size_t)i * 6;
  o[0] = (x * u[0] + y * u[3] + z * u[6]) * sx;
  o[1] = (x * u[1] + y * u[4] + z * u[7]) * sy;
  o[2] = (x * u[2] + y * u[5] + z * u[8]) * sz;
  o[3] = p[3];
  o[4] = p[4];
  o[5] = p[5];
}

// ---------------------------------------------------------------------------
// 2) farthest point sampling (scan semantics: idx[0]=0; first-max tie-break)
// ---------------------------------------------------------------------------
__global__ void __launch_bounds__(256) fps_kernel(const float* __restrict__ xyz,
                                                  int stride, int N, int npoint,
                                                  int* __restrict__ idx_out) {
  __shared__ float dist[NPTS];
  __shared__ float rv[256];
  __shared__ int   ri[256];
  int b = blockIdx.x, tid = threadIdx.x;
  const float* base = xyz + (size_t)b * N * stride;
  for (int j = tid; j < N; j += 256) dist[j] = 1e10f;
  int far = 0;
  if (tid == 0) idx_out[b * npoint] = 0;
  __syncthreads();
  for (int i = 1; i < npoint; ++i) {
    float cx = base[far * stride + 0];
    float cy = base[far * stride + 1];
    float cz = base[far * stride + 2];
    float bv = -1.f;
    int bi = 0;
    for (int j = tid; j < N; j += 256) {
      float dx = base[j * stride + 0] - cx;
      float dy = base[j * stride + 1] - cy;
      float dz = base[j * stride + 2] - cz;
      float d  = dx * dx + dy * dy + dz * dz;
      float nd = fminf(dist[j], d);
      dist[j] = nd;
      if (nd > bv) { bv = nd; bi = j; }
    }
    rv[tid] = bv;
    ri[tid] = bi;
    __syncthreads();
    for (int s = 128; s > 0; s >>= 1) {
      if (tid < s) {
        if (rv[tid + s] > rv[tid] ||
            (rv[tid + s] == rv[tid] && ri[tid + s] < ri[tid])) {
          rv[tid] = rv[tid + s];
          ri[tid] = ri[tid + s];
        }
      }
      __syncthreads();
    }
    far = ri[0];
    if (tid == 0) idx_out[b * npoint + i] = far;
    __syncthreads();
  }
}

// ---------------------------------------------------------------------------
// 3) gather centroid coordinates
// ---------------------------------------------------------------------------
__global__ void gather_xyz_kernel(const float* __restrict__ xyz, int stride, int N,
                                  const int* __restrict__ idx, int npoint,
                                  float* __restrict__ out) {
  int i = blockIdx.x * blockDim.x + threadIdx.x;
  if (i >= BATCH * npoint) return;
  int b = i / npoint;
  const float* src = xyz + ((size_t)b * N + idx[i]) * stride;
  out[i * 3 + 0] = src[0];
  out[i * 3 + 1] = src[1];
  out[i * 3 + 2] = src[2];
}

// ---------------------------------------------------------------------------
// 4) fused ball-query + 3-layer MLP (WMMA on LDS bf16) + max-pool
//    Feature gather: either f32 scalar path (sa1, 3 ch) or bf16 async
//    global->LDS bulk copy (sa2, 320 ch) via GLOBAL_LOAD_ASYNC_TO_LDS_B128.
// ---------------------------------------------------------------------------
static __device__ inline void mlp_layer(const bf16* in, int sin, int cinL,
                                        bf16* outb, int sout, int coutL,
                                        const bf16* Wp, const float* bias,
                                        const float* g, const float* be,
                                        int rows, int wave, int lane) {
  int mtiles = rows >> 4;
  int ntiles = coutL >> 4;
  int ktiles = (cinL + 31) >> 5;
  size_t bstep = (size_t)ntiles * 512;
  float rs = rsqrtf(1.f + EPSV);
  int arow = lane & 15;
  int aoff = (lane < 16) ? 0 : 8;
  for (int t = wave; t < mtiles * ntiles; t += 4) {
    int mt = t / ntiles, nt = t % ntiles;
    const bf16* pa = in + (size_t)(mt * 16 + arow) * sin + aoff;
    const bf16* pb = Wp + ((size_t)nt * 32 + lane) * 16;
    v8f acc = gemm_core(pa, pb, bstep, ktiles);
    int n  = nt * 16 + arow;
    float sc = g[n] * rs;
    float bi = bias[n];
    float bb = be[n];
    int mb = mt * 16 + aoff;
#pragma unroll
    for (int v = 0; v < 8; ++v) {
      float y = (acc[v] + bi) * sc + bb;     // bn_relu(x@W + bias)
      y = y > 0.f ? y : 0.f;
      outb[(size_t)(mb + v) * sout + n] = (bf16)y;
    }
  }
}

__global__ void __launch_bounds__(128) group_mlp_max_kernel(
    const float* __restrict__ xyz, int xstride, int Nsrc,
    const float* __restrict__ featsF,   // f32 feature source (or null)
    const bf16*  __restrict__ featsB,   // bf16 feature source (or null)
    int fstride, int cfeat,
    const float* __restrict__ new_xyz, int npoint, int K, float r2,
    const bf16* Wp0, const float* b0, const float* g0, const float* be0, int c0,
    const bf16* Wp1, const float* b1, const float* g1, const float* be1, int c1,
    const bf16* Wp2, const float* b2, const float* g2, const float* be2, int c2,
    int strideA, int strideB,
    bf16* __restrict__ out, int out_stride, int out_off) {
  extern __shared__ char smem_raw[];
  int*  nbr  = (int*)smem_raw;                           // 128 ints
  bf16* bufA = (bf16*)(smem_raw + 128 * sizeof(int));
  bf16* bufB = bufA + (size_t)K * strideA;

  int b = blockIdx.y, p = blockIdx.x;
  int tid = threadIdx.x, lane = tid & 31, wave = tid >> 5;
  const int cin = cfeat + 3;

  float cx = new_xyz[(b * npoint + p) * 3 + 0];
  float cy = new_xyz[(b * npoint + p) * 3 + 1];
  float cz = new_xyz[(b * npoint + p) * 3 + 2];
  const float* bx = xyz + (size_t)b * Nsrc * xstride;

  // --- ball query: first K in-radius points by ascending index, pad w/ first
  if (wave == 0) {
    int cnt = 0;
    for (int base = 0; base < Nsrc && cnt < K; base += 32) {
      int j = base + lane;
      bool inb = false;
      if (j < Nsrc) {
        float dx = bx[j * xstride + 0] - cx;
        float dy = bx[j * xstride + 1] - cy;
        float dz = bx[j * xstride + 2] - cz;
        inb = (dx * dx + dy * dy + dz * dz) <= r2;
      }
      unsigned m = (unsigned)__ballot(inb);   // wave32: low 32 bits
      int pos = cnt + __popc(m & ((1u << lane) - 1u));
      if (inb && pos < K) nbr[pos] = j;
      cnt += __popc(m);
    }
    if (cnt > K) cnt = K;
    for (int q = cnt + lane; q < K; q += 32) nbr[q] = nbr[0];
  }
  __syncthreads();

  int cinp = (cin + 31) & ~31;
  if (featsB) {
    // --- bf16 features: async global->LDS copy, 16B chunks (row = 640B)
    int chunks = (cfeat * (int)sizeof(bf16)) >> 4;   // 16-byte chunks per row
    for (int t = tid; t < K * chunks; t += 128) {
      int row = t / chunks, ch = t % chunks;
      const bf16* src = featsB + ((size_t)b * Nsrc + nbr[row]) * fstride;
      uint64_t ga = (uint64_t)(uintptr_t)src + (uint64_t)(ch << 4);
      uint32_t la = (uint32_t)(uintptr_t)(bufA + (size_t)row * strideA) +
                    (uint32_t)(ch << 4);
      asm volatile("global_load_async_to_lds_b128 %0, %1, off"
                   :: "v"(la), "v"(ga) : "memory");
    }
    // --- remaining columns: relative xyz + zero pad
    int rem = cinp - cfeat;
    for (int t = tid; t < K * rem; t += 128) {
      int j = t / rem, c = cfeat + t % rem;
      float v = 0.f;
      if (c < cin) {
        int d = c - cfeat;
        v = bx[nbr[j] * xstride + d] - ((d == 0) ? cx : (d == 1) ? cy : cz);
      }
      bufA[(size_t)j * strideA + c] = (bf16)v;
    }
    asm volatile("s_wait_asynccnt 0" ::: "memory");
  } else {
    // --- f32 features (sa1): scalar gather + convert
    for (int t = tid; t < K * cinp; t += 128) {
      int j = t / cinp, c = t % cinp;
      int nj = nbr[j];
      float v = 0.f;
      if (c < cfeat) {
        v = featsF[((size_t)b * Nsrc + nj) * fstride + c];
      } else if (c < cin) {
        int d = c - cfeat;
        v = bx[nj * xstride + d] - ((d == 0) ? cx : (d == 1) ? cy : cz);
      }
      bufA[(size_t)j * strideA + c] = (bf16)v;
    }
  }
  __syncthreads();

  mlp_layer(bufA, strideA, cin, bufB, strideB, c0, Wp0, b0, g0, be0, K, wave, lane);
  __syncthreads();
  mlp_layer(bufB, strideB, c0, bufA, strideA, c1, Wp1, b1, g1, be1, K, wave, lane);
  __syncthreads();
  mlp_layer(bufA, strideA, c1, bufB, strideB, c2, Wp2, b2, g2, be2, K, wave, lane);
  __syncthreads();

  // --- max-pool over K samples (bf16 out)
  for (int c = tid; c < c2; c += 128) {
    float mx = -1e30f;
    for (int j = 0; j < K; ++j) {
      float v = (float)bufB[(size_t)j * strideB + c];
      mx = v > mx ? v : mx;
    }
    out[((size_t)b * npoint + p) * out_stride + out_off + c] = (bf16)mx;
  }
}

// ---------------------------------------------------------------------------
// 5) WMMA GEMM on padded bf16 activations:
//    outb (bf16, padded) and/or outf (f32, bounds-guarded) = post(A @ W + bias)
//    mode 0: bias+bn+relu, mode 1: bias only
// ---------------------------------------------------------------------------
__global__ void __launch_bounds__(128) gemm_kernel(
    const bf16* __restrict__ A, int ldap, int ktiles,
    const bf16* __restrict__ Wp, int ntiles,
    const float* __restrict__ bias, const float* __restrict__ g,
    const float* __restrict__ be,
    bf16* __restrict__ outb, int ldob,
    float* __restrict__ outf, int ldof, int M, int Nn, int mode) {
  int lane = threadIdx.x & 31, wave = threadIdx.x >> 5;
  int mt = blockIdx.x;
  int nt = blockIdx.y * 4 + wave;
  if (nt >= ntiles) return;  // uniform per wave
  int arow = lane & 15;
  int aoff = (lane < 16) ? 0 : 8;
  const bf16* pa = A + (size_t)(mt * 16 + arow) * ldap + aoff;
  const bf16* pb = Wp + ((size_t)nt * 32 + lane) * 16;
  v8f acc = gemm_core(pa, pb, (size_t)ntiles * 512, ktiles);
  int n = nt * 16 + arow;
  float bi = (n < Nn) ? bias[n] : 0.f;
  float sc = 1.f, bb = 0.f;
  if (mode == 0 && n < Nn) { sc = g[n] * rsqrtf(1.f + EPSV); bb = be[n]; }
  int mb = mt * 16 + aoff;
#pragma unroll
  for (int v = 0; v < 8; ++v) {
    float y = acc[v] + bi;
    if (mode == 0) { y = y * sc + bb; y = y > 0.f ? y : 0.f; }
    if (outb) outb[(size_t)(mb + v) * ldob + n] = (bf16)y;
    if (outf && n < Nn && (mb + v) < M) outf[(size_t)(mb + v) * ldof + n] = y;
  }
}

// ---------------------------------------------------------------------------
// 6) misc small kernels
// ---------------------------------------------------------------------------
__global__ void pack_g3_kernel(const float* __restrict__ xyz2,
                               const bf16* __restrict__ f2,
                               bf16* __restrict__ g3p) {
  int i = blockIdx.x * blockDim.x + threadIdx.x;
  const int total = BATCH * 128 * 672;
  if (i >= total) return;
  int row = i / 672, c = i % 672;
  float v = 0.f;
  if (c < 3) v = xyz2[row * 3 + c];
  else if (c < 643) v = (float)f2[(size_t)row * 640 + (c - 3)];
  g3p[i] = (bf16)v;
}

__global__ void rowmax_kernel(const bf16* __restrict__ x, int ldx,
                              bf16* __restrict__ v, int ldv, int P, int C) {
  int i = blockIdx.x * blockDim.x + threadIdx.x;
  if (i >= BATCH * C) return;
  int b = i / C, c = i % C;
  float mx = -1e30f;
  for (int p = 0; p < P; ++p) {
    float t = (float)x[(size_t)(b * P + p) * ldx + c];
    mx = t > mx ? t : mx;
  }
  v[(size_t)b * ldv + c] = (bf16)mx;
}

__global__ void logsoftmax_kernel(const float* __restrict__ logits,
                                  float* __restrict__ out, int C) {
  __shared__ float red[64];
  int b = blockIdx.x, tid = threadIdx.x;
  float v = (tid < C) ? logits[b * C + tid] : -1e30f;
  red[tid] = v;
  __syncthreads();
  for (int s = 32; s > 0; s >>= 1) {
    if (tid < s) red[tid] = fmaxf(red[tid], red[tid + s]);
    __syncthreads();
  }
  float mx = red[0];
  __syncthreads();
  red[tid] = (tid < C) ? __expf(v - mx) : 0.f;
  __syncthreads();
  for (int s = 32; s > 0; s >>= 1) {
    if (tid < s) red[tid] += red[tid + s];
    __syncthreads();
  }
  float lse = __logf(red[0]) + mx;
  if (tid < C) out[b * C + tid] = v - lse;
}

__global__ void head_kernel(const float* __restrict__ SL /* 4 x B x 40 */,
                            const float* __restrict__ W1, const float* __restrict__ b1,
                            const float* __restrict__ W2, const float* __restrict__ b2,
                            float* __restrict__ out) {
  __shared__ float feat[40];
  __shared__ float h[64];
  int b = blockIdx.x, tid = threadIdx.x;  // 64 threads
  if (tid < 40) {
    float mx = -1e30f;
    for (int s = 0; s < 4; ++s) mx = fmaxf(mx, SL[(s * BATCH + b) * 40 + tid]);
    float sum = 0.f;
    for (int s = 0; s < 4; ++s) sum += __expf(SL[(s * BATCH + b) * 40 + tid] - mx);
    feat[tid] = __logf(sum) + mx;
  }
  __syncthreads();
  float acc = b1[tid];
  for (int c = 0; c < 40; ++c) acc += feat[c] * W1[c * 64 + tid];
  h[tid] = fmaxf(acc, 0.f);
  __syncthreads();
  if (tid == 0) {
    float o = b2[0];
    for (int t = 0; t < 64; ++t) o += h[t] * W2[t];
    out[b] = o;
  }
}

// ---------------------------------------------------------------------------
// host driver
// ---------------------------------------------------------------------------
struct Lyr { const float *W, *b, *g, *be; };

extern "C" void kernel_launch(void* const* d_in, const int* in_sizes, int n_in,
                              void* d_out, int out_size, void* d_ws, size_t ws_size,
                              hipStream_t stream) {
  (void)in_sizes; (void)n_in; (void)out_size; (void)ws_size;
  const float* P = (const float*)d_in[0];
  const float* U = (const float*)d_in[1];
  int pi = 2;
  auto nxt = [&]() { return (const float*)d_in[pi++]; };

  Lyr sa1L[3][3], sa2L[3][3], sa3L[3];
  for (int m = 0; m < 3; ++m)
    for (int l = 0; l < 3; ++l) { sa1L[m][l].W = nxt(); sa1L[m][l].b = nxt(); sa1L[m][l].g = nxt(); sa1L[m][l].be = nxt(); }
  for (int m = 0; m < 3; ++m)
    for (int l = 0; l < 3; ++l) { sa2L[m][l].W = nxt(); sa2L[m][l].b = nxt(); sa2L[m][l].g = nxt(); sa2L[m][l].be = nxt(); }
  for (int l = 0; l < 3; ++l) { sa3L[l].W = nxt(); sa3L[l].b = nxt(); sa3L[l].g = nxt(); sa3L[l].be = nxt(); }
  const float *fc1W = nxt(), *fc1b = nxt(), *fc1g = nxt(), *fc1be = nxt();
  const float *fc2W = nxt(), *fc2b = nxt(), *fc2g = nxt(), *fc2be = nxt();
  const float *fc3W = nxt(), *fc3b = nxt();
  const float *l1W = nxt(), *l1b = nxt();
  const float *l2W = nxt(), *l2b = nxt();

  // byte-based, 256B-aligned workspace carve-up
  char* wsb = (char*)d_ws;
  size_t off = 0;
  auto allocB = [&](size_t bytes) {
    void* p = wsb + off;
    off += (bytes + 255) & ~(size_t)255;
    return p;
  };
  float* aligned = (float*)allocB((size_t)BATCH * NPTS * 6 * 4);
  float* xyz1    = (float*)allocB((size_t)BATCH * 512 * 3 * 4);
  bf16*  f1b     = (bf16*)allocB((size_t)BATCH * 512 * 320 * 2);
  float* xyz2    = (float*)allocB((size_t)BATCH * 128 * 3 * 4);
  bf16*  f2b     = (bf16*)allocB((size_t)BATCH * 128 * 640 * 2);
  int*   idx1    = (int*)allocB((size_t)BATCH * 512 * 4);
  int*   idx2    = (int*)allocB((size_t)BATCH * 128 * 4);
  bf16*  g3p     = (bf16*)allocB((size_t)BATCH * 128 * 672 * 2);
  bf16*  s3a     = (bf16*)allocB((size_t)BATCH * 128 * 256 * 2);
  bf16*  s3b     = (bf16*)allocB((size_t)BATCH * 128 * 512 * 2);
  bf16*  s3c     = (bf16*)allocB((size_t)BATCH * 128 * 1024 * 2);
  bf16*  vfeatp  = (bf16*)allocB((size_t)16 * 1024 * 2);
  bf16*  h1p     = (bf16*)allocB((size_t)16 * 512 * 2);
  bf16*  h2p     = (bf16*)allocB((size_t)16 * 256 * 2);
  float* lgt     = (float*)allocB((size_t)BATCH * 40 * 4);
  float* SL      = (float*)allocB((size_t)4 * BATCH * 40 * 4);

  auto cdiv  = [](int a, int b) { return (a + b - 1) / b; };
  auto pad32 = [](int x) { return (x + 31) & ~31; };

  // ---- pre-pack all weights into WMMA B-fragment layout (bf16) ----
  auto pack = [&](const float* W, int Kd, int Nn) -> const bf16* {
    int kt = pad32(Kd) / 32, nt = ((Nn + 15) & ~15) / 16;
    size_t elems = (size_t)kt * nt * 512;
    bf16* o = (bf16*)allocB(elems * 2);
    pack_w_kernel<<<cdiv((int)elems, 256), 256, 0, stream>>>(W, Kd, Nn, o, kt, nt);
    return o;
  };
  const int sa1c[3][4] = {{6, 32, 32, 64}, {6, 64, 64, 128}, {6, 64, 96, 128}};
  const int sa2c[3][4] = {{323, 64, 64, 128}, {323, 128, 128, 256}, {323, 128, 128, 256}};
  const int sa3c[4]    = {643, 256, 512, 1024};
  const bf16 *sa1P[3][3], *sa2P[3][3], *sa3P[3];
  for (int m = 0; m < 3; ++m)
    for (int l = 0; l < 3; ++l) sa1P[m][l] = pack(sa1L[m][l].W, sa1c[m][l], sa1c[m][l + 1]);
  for (int m = 0; m < 3; ++m)
    for (int l = 0; l < 3; ++l) sa2P[m][l] = pack(sa2L[m][l].W, sa2c[m][l], sa2c[m][l + 1]);
  for (int l = 0; l < 3; ++l) sa3P[l] = pack(sa3L[l].W, sa3c[l], sa3c[l + 1]);
  const bf16* fc1P = pack(fc1W, 1024, 512);
  const bf16* fc2P = pack(fc2W, 512, 256);
  const bf16* fc3P = pack(fc3W, 256, 40);

  // allow >64KB dynamic LDS for the fused group kernel (max ~153 KB)
  hipFuncSetAttribute(reinterpret_cast<const void*>(&group_mlp_max_kernel),
                      hipFuncAttributeMaxDynamicSharedMemorySize, 160 * 1024);

  const float S[4][3] = {{1, 1, 1}, {1, -1, -1}, {-1, 1, -1}, {-1, -1, 1}};
  const float sa1r[3] = {0.1f, 0.2f, 0.4f};
  const float sa2r[3] = {0.2f, 0.4f, 0.8f};
  const int   sa1k[3] = {16, 32, 128};
  const int   sa2k[3] = {32, 64, 128};
  const int   sa1o[3] = {0, 64, 192};
  const int   sa2o[3] = {0, 128, 384};

  for (int s = 0; s < 4; ++s) {
    align_kernel<<<cdiv(BATCH * NPTS, 256), 256, 0, stream>>>(P, U, S[s][0], S[s][1], S[s][2], aligned);

    // ---- SA1 (N=2048 -> 512 centroids, MSG, f32 scalar feature path) ----
    fps_kernel<<<BATCH, 256, 0, stream>>>(aligned, 6, NPTS, 512, idx1);
    gather_xyz_kernel<<<cdiv(BATCH * 512, 256), 256, 0, stream>>>(aligned, 6, NPTS, idx1, 512, xyz1);
    for (int m = 0; m < 3; ++m) {
      const int* c = sa1c[m];
      int sA = pad32(c[0]) > pad32(c[2]) ? pad32(c[0]) : pad32(c[2]);
      int sB = pad32(c[1]) > pad32(c[3]) ? pad32(c[1]) : pad32(c[3]);
      size_t smem = 128 * sizeof(int) + (size_t)sa1k[m] * (sA + sB) * sizeof(bf16);
      group_mlp_max_kernel<<<dim3(512, BATCH), 128, smem, stream>>>(
          aligned, 6, NPTS, aligned + 3, nullptr, 6, 3, xyz1, 512, sa1k[m],
          sa1r[m] * sa1r[m],
          sa1P[m][0], sa1L[m][0].b, sa1L[m][0].g, sa1L[m][0].be, c[1],
          sa1P[m][1], sa1L[m][1].b, sa1L[m][1].g, sa1L[m][1].be, c[2],
          sa1P[m][2], sa1L[m][2].b, sa1L[m][2].g, sa1L[m][2].be, c[3],
          sA, sB, f1b, 320, sa1o[m]);
    }

    // ---- SA2 (512 -> 128 centroids, MSG, bf16 async feature path) ----
    fps_kernel<<<BATCH, 256, 0, stream>>>(xyz1, 3, 512, 128, idx2);
    gather_xyz_kernel<<<cdiv(BATCH * 128, 256), 256, 0, stream>>>(xyz1, 3, 512, idx2, 128, xyz2);
    for (int m = 0; m < 3; ++m) {
      const int* c = sa2c[m];
      int sA = pad32(c[0]) > pad32(c[2]) ? pad32(c[0]) : pad32(c[2]);
      int sB = pad32(c[1]) > pad32(c[3]) ? pad32(c[1]) : pad32(c[3]);
      size_t smem = 128 * sizeof(int) + (size_t)sa2k[m] * (sA + sB) * sizeof(bf16);
      group_mlp_max_kernel<<<dim3(128, BATCH), 128, smem, stream>>>(
          xyz1, 3, 512, nullptr, f1b, 320, 320, xyz2, 128, sa2k[m],
          sa2r[m] * sa2r[m],
          sa2P[m][0], sa2L[m][0].b, sa2L[m][0].g, sa2L[m][0].be, c[1],
          sa2P[m][1], sa2L[m][1].b, sa2L[m][1].g, sa2L[m][1].be, c[2],
          sa2P[m][2], sa2L[m][2].b, sa2L[m][2].g, sa2L[m][2].be, c[3],
          sA, sB, f2b, 640, sa2o[m]);
    }

    // ---- SA3 (1024 rows = B*128, 643 -> 256 -> 512 -> 1024, bf16 chain) ----
    pack_g3_kernel<<<cdiv(BATCH * 128 * 672, 256), 256, 0, stream>>>(xyz2, f2b, g3p);
    gemm_kernel<<<dim3(64, cdiv(16, 4)), 128, 0, stream>>>(
        g3p, 672, 21, sa3P[0], 16, sa3L[0].b, sa3L[0].g, sa3L[0].be,
        s3a, 256, nullptr, 0, 1024, 256, 0);
    gemm_kernel<<<dim3(64, cdiv(32, 4)), 128, 0, stream>>>(
        s3a, 256, 8, sa3P[1], 32, sa3L[1].b, sa3L[1].g, sa3L[1].be,
        s3b, 512, nullptr, 0, 1024, 512, 0);
    gemm_kernel<<<dim3(64, cdiv(64, 4)), 128, 0, stream>>>(
        s3b, 512, 16, sa3P[2], 64, sa3L[2].b, sa3L[2].g, sa3L[2].be,
        s3c, 1024, nullptr, 0, 1024, 1024, 0);
    rowmax_kernel<<<cdiv(BATCH * 1024, 256), 256, 0, stream>>>(s3c, 1024, vfeatp, 1024, 128, 1024);

    // ---- FC head (M=8 padded into one 16-row tile) ----
    gemm_kernel<<<dim3(1, cdiv(32, 4)), 128, 0, stream>>>(
        vfeatp, 1024, 32, fc1P, 32, fc1b, fc1g, fc1be,
        h1p, 512, nullptr, 0, BATCH, 512, 0);
    gemm_kernel<<<dim3(1, cdiv(16, 4)), 128, 0, stream>>>(
        h1p, 512, 16, fc2P, 16, fc2b, fc2g, fc2be,
        h2p, 256, nullptr, 0, BATCH, 256, 0);
    gemm_kernel<<<dim3(1, 1), 128, 0, stream>>>(
        h2p, 256, 8, fc3P, 3, fc3b, nullptr, nullptr,
        nullptr, 0, lgt, 40, BATCH, 40, 1);
    logsoftmax_kernel<<<BATCH, 64, 0, stream>>>(lgt, SL + (size_t)s * BATCH * 40, 40);
  }

  head_kernel<<<BATCH, 64, 0, stream>>>(SL, l1W, l1b, l2W, l2b, (float*)d_out);
}
